// TransformerEncoder_63127429317151
// MI455X (gfx1250) — matrix operand order repaired and checked by reading, not compiled, not run on hardware
//
#include <hip/hip_runtime.h>

// ---------------- CDNA5 WMMA types/helpers ----------------
typedef __attribute__((ext_vector_type(16))) _Float16 v16h;
typedef __attribute__((ext_vector_type(8)))  _Float16 v8h;
typedef __attribute__((ext_vector_type(8)))  float    v8f;

#define WMMA_F32_F16(a, b, c) \
  __builtin_amdgcn_wmma_f32_16x16x32_f16(false, (a), false, (b), (short)0, (c), false, false)

// A fragment: 16x32 f16 tile at (m_off, k_off) of row-major [rows x row_stride].
// Layout: lane&15 -> M row; lanes0-15 hold K 0..15, lanes16-31 hold K 16..31 (matches B).
__device__ __forceinline__ v16h load_a_frag(const _Float16* __restrict__ base, int row_stride,
                                            int m_off, int k_off, int lane) {
  int m  = m_off + (lane & 15);
  int kb = k_off + ((lane & 16) ? 16 : 0);
  const _Float16* p = base + (size_t)m * row_stride + kb;
  v8h lo = *(const v8h*)(p);       // K = kb .. kb+7
  v8h hi = *(const v8h*)(p + 8);   // K = kb+8 .. kb+15
  v16h r;
#pragma unroll
  for (int i = 0; i < 8; ++i) { r[i] = lo[i]; r[i + 8] = hi[i]; }
  return r;
}

// B fragment: 32x16 f16 (KxN) where memory holds B transposed: baseT is [N x row_stride(=K)],
// i.e. element B[k][n] = baseT[n][k]. Layout used here (same as A): lane&15 -> N; K packed along register,
// lanes0-15 K=0..15, lanes16-31 K=16..31 -> one contiguous 32B load per lane.
__device__ __forceinline__ v16h load_b_frag(const _Float16* __restrict__ baseT, int row_stride,
                                            int n_off, int k_off, int lane) {
  int n  = n_off + (lane & 15);
  int kb = k_off + ((lane & 16) ? 16 : 0);
  return *(const v16h*)(baseT + (size_t)n * row_stride + kb);
}

// C/D layout: VGPR r -> row (r + (lane>=16 ? 8 : 0)), col = lane&15.

// ---------------- constants ----------------
#define BB   16
#define NN   512
#define DIMC 768
#define MLP  3072
#define HH   12
#define DH   64
#define MROWS (BB * NN)   // 8192

// ---------------- weight convert: f32 [K,N] -> f16 [N,K] ----------------
__global__ void wconv_kernel(const float* __restrict__ W, _Float16* __restrict__ Wt,
                             int K, int N) {
  int idx = blockIdx.x * 256 + threadIdx.x;
  if (idx < K * N) {
    int k = idx / N, n = idx % N;
    Wt[(size_t)n * K + k] = (_Float16)W[idx];
  }
}

// ---------------- layernorm: f32 [8192,768] -> f16 [8192,768] ----------------
__global__ void ln_kernel(const float* __restrict__ x, const float* __restrict__ w,
                          const float* __restrict__ b, _Float16* __restrict__ out) {
  __shared__ float red[16];
  int row = blockIdx.x;
  const float* xr = x + (size_t)row * DIMC;
  int tid = threadIdx.x;
  float v0 = xr[tid], v1 = xr[tid + 256], v2 = xr[tid + 512];
  float s = v0 + v1 + v2;
  float s2 = v0 * v0 + v1 * v1 + v2 * v2;
#pragma unroll
  for (int m = 1; m < 32; m <<= 1) {
    s  += __shfl_xor(s, m, 32);
    s2 += __shfl_xor(s2, m, 32);
  }
  int wid = tid >> 5;
  if ((tid & 31) == 0) { red[wid] = s; red[8 + wid] = s2; }
  __syncthreads();
  float ts = 0.f, ts2 = 0.f;
#pragma unroll
  for (int i = 0; i < 8; ++i) { ts += red[i]; ts2 += red[8 + i]; }
  float mu  = ts * (1.0f / DIMC);
  float var = ts2 * (1.0f / DIMC) - mu * mu;
  float inv = rsqrtf(var + 1e-5f);
  _Float16* orow = out + (size_t)row * DIMC;
  orow[tid]       = (_Float16)((v0 - mu) * inv * w[tid]       + b[tid]);
  orow[tid + 256] = (_Float16)((v1 - mu) * inv * w[tid + 256] + b[tid + 256]);
  orow[tid + 512] = (_Float16)((v2 - mu) * inv * w[tid + 512] + b[tid + 512]);
}

// ---------------- generic WMMA GEMM ----------------
// C[M,N] = A[M,K](f16 row-major) x W[K,N] (given as Wt[N,K] f16) + bias, epilogue MODE.
// 8 waves/block arranged 4(M) x 2(N); each wave owns a 64x64 tile (4x4 WMMA tiles)
// -> block tile 256(M) x 128(N). K processed in steps of 64 with two fragment
// sets software-pipelined: both sets' loads issue before set-0's WMMAs, so
// set-1's loads overlap set-0's matrix work.
#define MODE_PLAIN 0  // f16 out row-major [M,N]
#define MODE_GELU  1  // f16 out row-major [M,N], exact gelu
#define MODE_BHND  2  // f16 out [B,H,N,DH]   (m=b*512+n_row, col=h*64+d)
#define MODE_BHDN  3  // f16 out [B,H,DH,N]
#define MODE_RESID 4  // f32 out[M,N] += acc+bias

template <int MODE>
__global__ __launch_bounds__(256) void gemm_kernel(const _Float16* __restrict__ A,
                                                   const _Float16* __restrict__ Wt,
                                                   const float* __restrict__ bias,
                                                   void* __restrict__ outp,
                                                   int M, int N, int K) {
  int lane = threadIdx.x & 31;
  int wave = threadIdx.x >> 5;
  int m0 = blockIdx.y * 256 + (wave & 3) * 64;
  int n0 = blockIdx.x * 128 + (wave >> 2) * 64;

  v8f acc[4][4];
#pragma unroll
  for (int i = 0; i < 4; ++i)
#pragma unroll
    for (int j = 0; j < 4; ++j) acc[i][j] = (v8f){0.f, 0.f, 0.f, 0.f, 0.f, 0.f, 0.f, 0.f};

  for (int k = 0; k < K; k += 64) {
    v16h a0[4], b0[4], a1[4], b1[4];
    // issue both K-steps' loads before any WMMA consumes set 0
#pragma unroll
    for (int i = 0; i < 4; ++i) a0[i] = load_a_frag(A, K, m0 + 16 * i, k, lane);
#pragma unroll
    for (int j = 0; j < 4; ++j) b0[j] = load_b_frag(Wt, K, n0 + 16 * j, k, lane);
#pragma unroll
    for (int i = 0; i < 4; ++i) a1[i] = load_a_frag(A, K, m0 + 16 * i, k + 32, lane);
#pragma unroll
    for (int j = 0; j < 4; ++j) b1[j] = load_b_frag(Wt, K, n0 + 16 * j, k + 32, lane);
#pragma unroll
    for (int i = 0; i < 4; ++i)
#pragma unroll
      for (int j = 0; j < 4; ++j) acc[i][j] = WMMA_F32_F16(a0[i], b0[j], acc[i][j]);
#pragma unroll
    for (int i = 0; i < 4; ++i)
#pragma unroll
      for (int j = 0; j < 4; ++j) acc[i][j] = WMMA_F32_F16(a1[i], b1[j], acc[i][j]);
  }

#pragma unroll
  for (int j = 0; j < 4; ++j) {
    int col = n0 + 16 * j + (lane & 15);
    float bv = bias[col];
#pragma unroll
    for (int i = 0; i < 4; ++i) {
      int rbase = m0 + 16 * i + ((lane & 16) ? 8 : 0);
#pragma unroll
      for (int r = 0; r < 8; ++r) {
        int row = rbase + r;
        float val = acc[i][j][r] + bv;
        if constexpr (MODE == MODE_PLAIN) {
          ((_Float16*)outp)[(size_t)row * N + col] = (_Float16)val;
        } else if constexpr (MODE == MODE_GELU) {
          float g = 0.5f * val * (1.0f + erff(val * 0.70710678118654752f));
          ((_Float16*)outp)[(size_t)row * N + col] = (_Float16)g;
        } else if constexpr (MODE == MODE_BHND) {
          int bb = row >> 9, nr = row & 511;
          int hh = col >> 6, dd = col & 63;
          size_t idx = (((size_t)bb * HH + hh) * NN + nr) * DH + dd;
          ((_Float16*)outp)[idx] = (_Float16)val;
        } else if constexpr (MODE == MODE_BHDN) {
          int bb = row >> 9, nr = row & 511;
          int hh = col >> 6, dd = col & 63;
          size_t idx = (((size_t)bb * HH + hh) * DH + dd) * NN + nr;
          ((_Float16*)outp)[idx] = (_Float16)val;
        } else {  // MODE_RESID
          float* o = (float*)outp;
          o[(size_t)row * N + col] += val;
        }
      }
    }
  }
}

// ---------------- flash attention: Q[B,H,N,DH], K[B,H,N,DH], Vt[B,H,DH,N] -> xcur += O ----------------
__global__ __launch_bounds__(256) void attn_kernel(const _Float16* __restrict__ Q,
                                                   const _Float16* __restrict__ Km,
                                                   const _Float16* __restrict__ Vt,
                                                   float* __restrict__ xcur) {
  __shared__ __align__(64) _Float16 pbuf[8][16 * 32];
  int bh = blockIdx.x;                 // b*12 + h
  int b = bh / HH, h = bh % HH;
  const _Float16* Qb = Q  + (size_t)bh * NN * DH;
  const _Float16* Kb = Km + (size_t)bh * NN * DH;
  const _Float16* Vb = Vt + (size_t)bh * DH * NN;
  float* xout = xcur + (size_t)b * NN * DIMC + h * DH;

  int lane = threadIdx.x & 31;
  int wave = threadIdx.x >> 5;
  _Float16* pw = pbuf[wave];

  for (int it = 0; it < 4; ++it) {
    int q0 = (it * 8 + wave) * 16;
    v16h aQ0 = load_a_frag(Qb, DH, q0, 0, lane);
    v16h aQ1 = load_a_frag(Qb, DH, q0, 32, lane);

    v8f accO[4];
#pragma unroll
    for (int t = 0; t < 4; ++t) accO[t] = (v8f){0.f, 0.f, 0.f, 0.f, 0.f, 0.f, 0.f, 0.f};
    float mrow[8], lrow[8];
#pragma unroll
    for (int r = 0; r < 8; ++r) { mrow[r] = -1e30f; lrow[r] = 0.f; }

    for (int kt = 0; kt < NN; kt += 32) {
      // S tile 16x32: cols kt..kt+31
      v16h bk00 = load_b_frag(Kb, DH, kt,      0,  lane);
      v16h bk01 = load_b_frag(Kb, DH, kt,      32, lane);
      v16h bk10 = load_b_frag(Kb, DH, kt + 16, 0,  lane);
      v16h bk11 = load_b_frag(Kb, DH, kt + 16, 32, lane);
      v8f s0 = (v8f){0.f, 0.f, 0.f, 0.f, 0.f, 0.f, 0.f, 0.f};
      v8f s1 = s0;
      s0 = WMMA_F32_F16(aQ0, bk00, s0);
      s0 = WMMA_F32_F16(aQ1, bk01, s0);
      s1 = WMMA_F32_F16(aQ0, bk10, s1);
      s1 = WMMA_F32_F16(aQ1, bk11, s1);

      const float sc = 0.125f;  // 1/sqrt(64)
#pragma unroll
      for (int r = 0; r < 8; ++r) {
        float e0 = s0[r] * sc, e1 = s1[r] * sc;
        float tm = fmaxf(e0, e1);
#pragma unroll
        for (int m = 1; m < 16; m <<= 1) tm = fmaxf(tm, __shfl_xor(tm, m, 32));
        float nm = fmaxf(mrow[r], tm);
        float alpha = __expf(mrow[r] - nm);
        float p0 = __expf(e0 - nm);
        float p1 = __expf(e1 - nm);
        float rs = p0 + p1;
#pragma unroll
        for (int m = 1; m < 16; m <<= 1) rs += __shfl_xor(rs, m, 32);
        lrow[r] = lrow[r] * alpha + rs;
        mrow[r] = nm;
#pragma unroll
        for (int t = 0; t < 4; ++t) accO[t][r] *= alpha;
        s0[r] = p0;
        s1[r] = p1;
      }

      // P (16x32, C-layout) -> LDS row-major -> A fragment
      {
        int prow = (lane & 16) ? 8 : 0;
        int pcol = lane & 15;
#pragma unroll
        for (int r = 0; r < 8; ++r) {
          pw[(prow + r) * 32 + pcol]      = (_Float16)s0[r];
          pw[(prow + r) * 32 + 16 + pcol] = (_Float16)s1[r];
        }
      }
      asm volatile("s_wait_dscnt 0" ::: "memory");
      v16h aP = load_a_frag(pw, 32, 0, 0, lane);

#pragma unroll
      for (int t = 0; t < 4; ++t) {
        v16h bv = load_b_frag(Vb, NN, t * 16, kt, lane);  // V[kt+k][d] = Vt[d][kt+k]
        accO[t] = WMMA_F32_F16(aP, bv, accO[t]);
      }
    }

    // epilogue: O /= l, residual-add into x
    float linv[8];
#pragma unroll
    for (int r = 0; r < 8; ++r) linv[r] = 1.0f / lrow[r];
#pragma unroll
    for (int t = 0; t < 4; ++t) {
#pragma unroll
      for (int r = 0; r < 8; ++r) {
        int row = q0 + r + ((lane & 16) ? 8 : 0);
        int col = t * 16 + (lane & 15);
        xout[(size_t)row * DIMC + col] += accO[t][r] * linv[r];
      }
    }
  }
}

// ---------------- host launch ----------------
extern "C" void kernel_launch(void* const* d_in, const int* in_sizes, int n_in,
                              void* d_out, int out_size, void* d_ws, size_t ws_size,
                              hipStream_t stream) {
  const float* x_in = (const float*)d_in[0];
  const float* ln_w = (const float*)d_in[1];
  const float* ln_b = (const float*)d_in[2];
  const float* wq = (const float*)d_in[3];
  const float* bq = (const float*)d_in[4];
  const float* wk = (const float*)d_in[5];
  const float* bk = (const float*)d_in[6];
  const float* wv = (const float*)d_in[7];
  const float* bv = (const float*)d_in[8];
  const float* w1 = (const float*)d_in[9];
  const float* b1 = (const float*)d_in[10];
  const float* w2 = (const float*)d_in[11];
  const float* b2 = (const float*)d_in[12];

  char* ws = (char*)d_ws;
  size_t off = 0;
  auto alloc = [&](size_t bytes) {
    size_t o = off;
    off += (bytes + 255) & ~(size_t)255;
    return o;
  };
  _Float16* xn  = (_Float16*)(ws + alloc((size_t)MROWS * DIMC * 2));
  _Float16* qB  = (_Float16*)(ws + alloc((size_t)MROWS * DIMC * 2));
  _Float16* kB  = (_Float16*)(ws + alloc((size_t)MROWS * DIMC * 2));
  _Float16* vtB = (_Float16*)(ws + alloc((size_t)MROWS * DIMC * 2));
  _Float16* h1  = (_Float16*)(ws + alloc((size_t)MROWS * MLP * 2));
  _Float16* wqT = (_Float16*)(ws + alloc((size_t)DIMC * DIMC * 2));
  _Float16* wkT = (_Float16*)(ws + alloc((size_t)DIMC * DIMC * 2));
  _Float16* wvT = (_Float16*)(ws + alloc((size_t)DIMC * DIMC * 2));
  _Float16* w1T = (_Float16*)(ws + alloc((size_t)DIMC * MLP * 2));   // [MLP, DIMC]
  _Float16* w2T = (_Float16*)(ws + alloc((size_t)MLP * DIMC * 2));   // [DIMC, MLP]

  float* xcur = (float*)d_out;
  hipMemcpyAsync(xcur, x_in, (size_t)MROWS * DIMC * sizeof(float),
                 hipMemcpyDeviceToDevice, stream);

  // one-time weight conversion (f32 [K,N] -> f16 [N,K])
  {
    int n768 = DIMC * DIMC, nBig = DIMC * MLP;
    wconv_kernel<<<(n768 + 255) / 256, 256, 0, stream>>>(wq, wqT, DIMC, DIMC);
    wconv_kernel<<<(n768 + 255) / 256, 256, 0, stream>>>(wk, wkT, DIMC, DIMC);
    wconv_kernel<<<(n768 + 255) / 256, 256, 0, stream>>>(wv, wvT, DIMC, DIMC);
    wconv_kernel<<<(nBig + 255) / 256, 256, 0, stream>>>(w1, w1T, DIMC, MLP);
    wconv_kernel<<<(nBig + 255) / 256, 256, 0, stream>>>(w2, w2T, MLP, DIMC);
  }

  dim3 blk(256);
  dim3 gQKV(DIMC / 128, MROWS / 256);  // (6, 32)
  dim3 gMLP1(MLP / 128, MROWS / 256);  // (24, 32)

  for (int layer = 0; layer < 12; ++layer) {
    ln_kernel<<<MROWS, blk, 0, stream>>>(xcur, ln_w, ln_b, xn);
    gemm_kernel<MODE_BHND><<<gQKV, blk, 0, stream>>>(xn, wqT, bq, qB, MROWS, DIMC, DIMC);
    gemm_kernel<MODE_BHND><<<gQKV, blk, 0, stream>>>(xn, wkT, bk, kB, MROWS, DIMC, DIMC);
    gemm_kernel<MODE_BHDN><<<gQKV, blk, 0, stream>>>(xn, wvT, bv, vtB, MROWS, DIMC, DIMC);
    attn_kernel<<<BB * HH, blk, 0, stream>>>(qB, kB, vtB, xcur);
    ln_kernel<<<MROWS, blk, 0, stream>>>(xcur, ln_w, ln_b, xn);
    gemm_kernel<MODE_GELU><<<gMLP1, blk, 0, stream>>>(xn, w1T, b1, h1, MROWS, MLP, DIMC);
    gemm_kernel<MODE_RESID><<<gQKV, blk, 0, stream>>>(h1, w2T, b2, xcur, MROWS, DIMC, MLP);
  }
}
